// SimilarityGuidedSampling_21964462752514
// MI455X (gfx1250) — matrix operand order repaired and checked by reading, not compiled, not run on hardware
//
#include <hip/hip_runtime.h>
#include <hip/hip_bf16.h>

typedef __attribute__((ext_vector_type(16))) _Float16 v16h;
typedef __attribute__((ext_vector_type(8)))  float    v8f;

#define B_  8
#define C_  256
#define T_  32
#define HW_ 784         // 28*28
#define HW4_ 196        // 784/4 (float4)
#define NB_ 4
#define HID_ 512
#define EMB_ 8
#define INTERVAL_SCALE (1.0f / 10.0f)   // 1/(4*(NUM_BINS-2)+2)

// ---------------------------------------------------------------------------
// Kernel 1: spatial mean pool. One wave per (b,c,t). Coalesced float4 loads,
// wave32 shuffle reduction, write f16 A-matrix pooled[bt][c].
// ---------------------------------------------------------------------------
__global__ void sgs_pool(const float* __restrict__ x, _Float16* __restrict__ pooled) {
    const int gtid = blockIdx.x * blockDim.x + threadIdx.x;
    const int wid  = gtid >> 5;            // (b*C + c)*T + t, matches x layout
    const int lane = threadIdx.x & 31;
    const float4* src = (const float4*)(x + (size_t)wid * HW_);
    float s = 0.0f;
    for (int v = lane; v < HW4_; v += 32) {
        float4 q = src[v];
        s += (q.x + q.y) + (q.z + q.w);
    }
    #pragma unroll
    for (int off = 16; off > 0; off >>= 1) s += __shfl_xor(s, off, 32);
    if (lane == 0) {
        const int t = wid & (T_ - 1);
        const int c = (wid / T_) & (C_ - 1);
        const int b = wid / (T_ * C_);
        pooled[(b * T_ + t) * C_ + c] = (_Float16)(s * (1.0f / (float)HW_));
    }
}

// ---------------------------------------------------------------------------
// Kernel 2: w1[HID][C] fp32 -> f16 pre-swizzled into per-lane WMMA B-fragment
// layout. Packed index: ((tn*8 + kblk)*32 + lane)*16 + i  holds
// B[k][n] with n = tn*16 + (lane&15), k = kblk*32 + (lane>>4)*8 + (i&7) + (i>>3)*16,
// i.e. B[k][n] = w1[n][k]. Each lane's fragment is then one contiguous 32B run.
// ---------------------------------------------------------------------------
__global__ void sgs_cvt_w1(const float* __restrict__ w1, _Float16* __restrict__ w1p) {
    const int idx  = blockIdx.x * blockDim.x + threadIdx.x;  // 0 .. C_*HID_-1
    const int i    = idx & 15;
    const int lane = (idx >> 4) & 31;
    const int kblk = (idx >> 9) & 7;          // C_/32 = 8 K blocks
    const int tn   = idx >> 12;               // HID_/16 = 32 N tiles
    const int n = tn * 16 + (lane & 15);      // HIDDEN index j
    const int k = kblk * 32 + (lane >> 4) * 8 + (i & 7) + (i >> 3) * 16;  // C index c
    w1p[idx] = (_Float16)w1[n * C_ + k];
}

// ---------------------------------------------------------------------------
// Kernel 3: encoder GEMM  h[BT,HID] = hswish(pooled[BT,C] @ w1^T + b1)
// One wave per 16x16 D tile, K-loop of 8 x v_wmma_f32_16x16x32_f16.
// A: two contiguous 16B runs per lane; B: one contiguous 32B fragment per lane.
// ---------------------------------------------------------------------------
__global__ void sgs_gemm1_wmma(const _Float16* __restrict__ Apool,  // [256][256]
                               const _Float16* __restrict__ Bpack,  // packed fragments
                               const float*    __restrict__ b1,
                               float*          __restrict__ hbuf) { // [256][512]
    const int tile = blockIdx.x;          // 0..511
    const int tn   = tile & 31;           // 512/16 N tiles
    const int tm   = tile >> 5;           // 256/16 M tiles
    const int lane = threadIdx.x & 31;
    const int half = lane >> 4;           // 0: lanes 0-15, 1: lanes 16-31
    const int l15  = lane & 15;
    const int m0 = tm * 16, n0 = tn * 16;

    const _Float16* arow  = Apool + (m0 + l15) * C_;
    const _Float16* bbase = Bpack + ((size_t)tn * 8 * 32 + lane) * 16;

    v8f acc = {};
    for (int kblk = 0; kblk < 8; ++kblk) {
        const int kk = kblk * 32;
        // A fragment: 16x32 f16. VGPR0-3: K = kk + half*8 + {0..7};
        // VGPR4-7: K = kk + half*8 + 16 + {0..7}. Both runs contiguous (16B).
        v16h a;
        const int kA = kk + half * 8;
        #pragma unroll
        for (int i = 0; i < 8; ++i) {
            a[i]     = arow[kA + i];
            a[i + 8] = arow[kA + 16 + i];
        }
        // B fragment: pre-swizzled, one contiguous 32B load per lane.
        const v16h bfrag = *(const v16h*)(bbase + (size_t)kblk * 32 * 16);

        acc = __builtin_amdgcn_wmma_f32_16x16x32_f16(
            /*neg_a=*/false, a, /*neg_b=*/false, bfrag,
            /*c_mod=*/(short)0, acc, /*reuse_a=*/false, /*reuse_b=*/false);
    }
    // D layout: lane l, VGPR r -> (M = m0 + r + 8*half, N = n0 + l15)
    const int n = n0 + l15;
    const float bias = b1[n];
    #pragma unroll
    for (int r = 0; r < 8; ++r) {
        const int m = m0 + r + half * 8;
        float h = acc[r] + bias;
        float cl = fminf(fmaxf(h + 3.0f, 0.0f), 6.0f);
        hbuf[m * HID_ + n] = h * cl * (1.0f / 6.0f);
    }
}

// ---------------------------------------------------------------------------
// Kernel 4: second (tiny) projection, norms, per-batch min/max, bin coeffs.
// One wave per batch; lane = t.
// ---------------------------------------------------------------------------
__global__ void sgs_coeff(const float* __restrict__ hbuf,   // [256][512]
                          const float* __restrict__ w2,     // [8][512]
                          const float* __restrict__ b2,     // [8]
                          float*       __restrict__ coeff) {// [B][T][NB]
    const int b = blockIdx.x;     // 0..7
    const int t = threadIdx.x;    // 0..31
    const float* hv = hbuf + (b * T_ + t) * HID_;

    float norm = 0.0f;
    #pragma unroll
    for (int e = 0; e < EMB_; ++e) {
        const float* wr = w2 + e * HID_;
        float s = b2[e];
        for (int j = 0; j < HID_; ++j) s = fmaf(wr[j], hv[j], s);
        norm = fmaf(s, s, norm);
    }

    float mn = norm, mx = norm;
    #pragma unroll
    for (int off = 16; off > 0; off >>= 1) {
        mn = fminf(mn, __shfl_xor(mn, off, 32));
        mx = fmaxf(mx, __shfl_xor(mx, off, 32));
    }
    const float gamma = INTERVAL_SCALE * (mx - mn);
    const float inv2g = 0.5f / gamma;

    float un[NB_];
    #pragma unroll
    for (int n = 0; n < NB_; ++n) {
        const float center = mn + gamma * (4.0f * (float)n - 1.0f);
        un[n] = fmaxf(1.0f - fabsf(norm - center) * inv2g, 0.0f);
    }
    #pragma unroll
    for (int n = 0; n < NB_; ++n) {
        float s = un[n];
        #pragma unroll
        for (int off = 16; off > 0; off >>= 1) s += __shfl_xor(s, off, 32);
        const float scale = (s > 0.0f) ? (1.0f / s) : 1.0f;
        coeff[(b * T_ + t) * NB_ + n] = scale * un[n];
    }
}

// ---------------------------------------------------------------------------
// Kernel 5: out[b,c,n,h,w] = sum_t x[b,c,t,h,w] * coeff[b,t,n]
// One block per (b,c); coeff[b] in LDS; float4 vectorized; 16 FMAs / 16B load.
// ---------------------------------------------------------------------------
__global__ void sgs_weighted_pool(const float* __restrict__ x,      // [B,C,T,784]
                                  const float* __restrict__ coeff,  // [B,T,NB]
                                  float*       __restrict__ out) {  // [B,C,NB,784]
    const int bc = blockIdx.x;            // b*C + c
    const int b  = bc >> 8;               // /C_
    __shared__ float sc[T_ * NB_];
    if (threadIdx.x < T_ * NB_) sc[threadIdx.x] = coeff[b * T_ * NB_ + threadIdx.x];
    __syncthreads();

    const float4* xb = (const float4*)(x + (size_t)bc * T_ * HW_);
    float4*       ob = (float4*)(out + (size_t)bc * NB_ * HW_);

    for (int p = threadIdx.x; p < HW4_; p += blockDim.x) {
        float4 a0 = {0.f,0.f,0.f,0.f}, a1 = a0, a2 = a0, a3 = a0;
        #pragma unroll 4
        for (int t = 0; t < T_; ++t) {
            const float4 xv = xb[t * HW4_ + p];
            const float c0 = sc[t*NB_+0], c1 = sc[t*NB_+1], c2 = sc[t*NB_+2], c3 = sc[t*NB_+3];
            a0.x = fmaf(xv.x, c0, a0.x); a0.y = fmaf(xv.y, c0, a0.y);
            a0.z = fmaf(xv.z, c0, a0.z); a0.w = fmaf(xv.w, c0, a0.w);
            a1.x = fmaf(xv.x, c1, a1.x); a1.y = fmaf(xv.y, c1, a1.y);
            a1.z = fmaf(xv.z, c1, a1.z); a1.w = fmaf(xv.w, c1, a1.w);
            a2.x = fmaf(xv.x, c2, a2.x); a2.y = fmaf(xv.y, c2, a2.y);
            a2.z = fmaf(xv.z, c2, a2.z); a2.w = fmaf(xv.w, c2, a2.w);
            a3.x = fmaf(xv.x, c3, a3.x); a3.y = fmaf(xv.y, c3, a3.y);
            a3.z = fmaf(xv.z, c3, a3.z); a3.w = fmaf(xv.w, c3, a3.w);
        }
        ob[0 * HW4_ + p] = a0;
        ob[1 * HW4_ + p] = a1;
        ob[2 * HW4_ + p] = a2;
        ob[3 * HW4_ + p] = a3;
    }
}

// ---------------------------------------------------------------------------
extern "C" void kernel_launch(void* const* d_in, const int* in_sizes, int n_in,
                              void* d_out, int out_size, void* d_ws, size_t ws_size,
                              hipStream_t stream) {
    const float* x  = (const float*)d_in[0];   // [B,C,T,H,W]
    const float* w1 = (const float*)d_in[1];   // [HID,C]
    const float* b1 = (const float*)d_in[2];   // [HID]
    const float* w2 = (const float*)d_in[3];   // [EMB,HID]
    const float* b2 = (const float*)d_in[4];   // [EMB]
    float* out = (float*)d_out;                // [B,C,NB,H,W]

    // workspace carve-up (256B aligned)
    char* ws = (char*)d_ws;
    _Float16* pooled = (_Float16*)ws;                               // 256*256*2 = 128KB
    _Float16* w1p    = (_Float16*)(ws + 131072);                    // 256*512*2 = 256KB
    float*    hbuf   = (float*)(ws + 131072 + 262144);              // 256*512*4 = 512KB
    float*    coeff  = (float*)(ws + 131072 + 262144 + 524288);     // 1024*4    =   4KB

    // 1) spatial mean pool: 65536 waves (8 per 256-thread block)
    sgs_pool<<<(B_ * C_ * T_) / 8, 256, 0, stream>>>(x, pooled);

    // 2) w1 -> f16, pre-swizzled into WMMA B-fragment layout
    sgs_cvt_w1<<<(C_ * HID_) / 256, 256, 0, stream>>>(w1, w1p);

    // 3) WMMA encoder GEMM + bias + hswish: 16x16 tiles, one wave each
    sgs_gemm1_wmma<<<(C_ / 16) * (HID_ / 16), 32, 0, stream>>>(pooled, w1p, b1, hbuf);

    // 4) projection + norms + bin coefficients: one wave per batch
    sgs_coeff<<<B_, 32, 0, stream>>>(hbuf, w2, b2, coeff);

    // 5) weighted temporal pooling: one block per (b,c)
    sgs_weighted_pool<<<B_ * C_, 256, 0, stream>>>(x, coeff, out);
}
